// SparseAttention_38079180046463
// MI455X (gfx1250) — matrix operand order repaired and checked by reading, not compiled, not run on hardware
//
#include <hip/hip_runtime.h>

typedef _Float16 v8h  __attribute__((ext_vector_type(8)));
typedef _Float16 v16h __attribute__((ext_vector_type(16)));
typedef float    v8f  __attribute__((ext_vector_type(8)));

#define TOKENS 65536   // B*N
#define EDIM   256
#define HD     128
#define BSZ    8

// ---------------- workspace layout (bytes) ----------------
// effective transposed weights (f16, [out_col][k]):
static constexpr size_t OFF_WQT = 0;                    // 256*256*2 = 128K
static constexpr size_t OFF_WKT = 131072;
static constexpr size_t OFF_WVT = 262144;
static constexpr size_t OFF_BQ  = 393216;               // 256 f32
static constexpr size_t OFF_BK  = 394240;
static constexpr size_t OFF_BV  = 395264;
static constexpr size_t OFF_Q   = 524288;               // f16 [65536][256] = 32M
static constexpr size_t OFF_K   = OFF_Q + (size_t)TOKENS*EDIM*2;
static constexpr size_t OFF_V   = OFF_K + (size_t)TOKENS*EDIM*2;
// total ~96.6 MB

// ------------------------------------------------------------------
// Kernel 1: fold P into Wq/Wk, transpose into [col][k] f16 layout.
//   WqT_eff[j][e] = sum_d Wq[(j/128)*128+d][e] * P[d][j%128]
//   WvT[j][e]     = Wv[j][e]
// ------------------------------------------------------------------
__global__ __launch_bounds__(256) void prep_kernel(
    const float* __restrict__ Wq, const float* __restrict__ bq,
    const float* __restrict__ Wk, const float* __restrict__ bk,
    const float* __restrict__ Wv, const float* __restrict__ bv,
    const float* __restrict__ P,
    _Float16* __restrict__ WqT, _Float16* __restrict__ WkT,
    _Float16* __restrict__ WvT,
    float* __restrict__ bQ, float* __restrict__ bK, float* __restrict__ bV)
{
    int t = blockIdx.x * blockDim.x + threadIdx.x;
    if (t < 65536) {                       // WqT_eff
        int j = t >> 8, e = t & 255;
        int h = j >> 7, i = j & 127;
        float s = 0.f;
        for (int d = 0; d < 128; ++d)
            s += Wq[(size_t)(h * 128 + d) * 256 + e] * P[d * 128 + i];
        WqT[(size_t)j * 256 + e] = (_Float16)s;
    } else if (t < 131072) {               // WkT_eff
        int tt = t - 65536;
        int j = tt >> 8, e = tt & 255;
        int h = j >> 7, i = j & 127;
        float s = 0.f;
        for (int d = 0; d < 128; ++d)
            s += Wk[(size_t)(h * 128 + d) * 256 + e] * P[d * 128 + i];
        WkT[(size_t)j * 256 + e] = (_Float16)s;
    } else if (t < 196608) {               // WvT (plain transpose-by-convention)
        int idx = t - 131072;
        WvT[idx] = (_Float16)Wv[idx];
    } else if (t < 197120) {               // effective q/k biases (bias @ P)
        int idx = t - 196608;
        int which = idx >> 8, j = idx & 255;
        int h = j >> 7, i = j & 127;
        const float* b = which ? bk : bq;
        float s = 0.f;
        for (int d = 0; d < 128; ++d)
            s += b[h * 128 + d] * P[d * 128 + i];
        (which ? bK : bQ)[j] = s;
    } else if (t < 197376) {               // v bias passthrough
        int j = t - 197120;
        bV[j] = bv[j];
    }
}

// ------------------------------------------------------------------
// Kernel 2: fused QKV GEMM.  Grid = 1024 WGs, 512 thr (16 waves).
// WG handles 64 tokens x 256 cols for all three matrices.
// Wave w owns 16-col strip cs=w.  x tile staged f16 in LDS.
// ------------------------------------------------------------------
#define ROWS 64
#define LDSP 264   // 256 + 8 half pad (breaks 512B-stride bank conflicts)

__global__ __launch_bounds__(512) void qkv_kernel(
    const float* __restrict__ x,
    const _Float16* __restrict__ WqT, const _Float16* __restrict__ WkT,
    const _Float16* __restrict__ WvT,
    const float* __restrict__ bQ, const float* __restrict__ bK,
    const float* __restrict__ bV,
    _Float16* __restrict__ qws, _Float16* __restrict__ kws,
    _Float16* __restrict__ vws)
{
    __shared__ _Float16 ldsx[ROWS * LDSP];   // 33792 B

    const int tid     = threadIdx.x;
    const int rowbase = blockIdx.x * ROWS;

    // ---- cooperative load x (f32) -> LDS (f16) ----
    #pragma unroll
    for (int i = 0; i < 8; ++i) {
        int flat4 = tid + i * 512;           // 4096 float4s total
        int row   = flat4 >> 6;              // 64 float4 per row
        int c4    = flat4 & 63;
        float4 f = *(const float4*)(x + (size_t)(rowbase + row) * EDIM + c4 * 4);
        _Float16* dst = ldsx + row * LDSP + c4 * 4;
        dst[0] = (_Float16)f.x; dst[1] = (_Float16)f.y;
        dst[2] = (_Float16)f.z; dst[3] = (_Float16)f.w;
    }
    __syncthreads();

    const int lane = tid & 31;
    const int cs   = tid >> 5;               // column strip 0..15
    const int col  = cs * 16 + (lane & 15);
    const int bsel = (lane >> 4) << 4;       // +16 halfs for upper half-wave (B frag)
    const int ab   = (lane >> 4) << 4;       // +16 halfs for upper half-wave (A frag)
    const int am   = lane & 15;              // A row within 16-row tile

    v8f accq[4] = {}, acck[4] = {}, accv[4] = {};

    #pragma unroll
    for (int ks = 0; ks < 8; ++ks) {
        const int k0 = ks * 32;
        // B fragments: one aligned 32B load each (32x16 f16 B layout)
        v16h bqf = *(const v16h*)(WqT + (size_t)col * 256 + k0 + bsel);
        v16h bkf = *(const v16h*)(WkT + (size_t)col * 256 + k0 + bsel);
        v16h bvf = *(const v16h*)(WvT + (size_t)col * 256 + k0 + bsel);
        #pragma unroll
        for (int rt = 0; rt < 4; ++rt) {
            // A fragment (16x32 f16 layout): Ks {b..b+15}
            const _Float16* ap = ldsx + (rt * 16 + am) * LDSP + k0 + ab;
            v8h lo = *(const v8h*)(ap);
            v8h hi = *(const v8h*)(ap + 8);
            v16h a = __builtin_shufflevector(lo, hi, 0,1,2,3,4,5,6,7,
                                                     8,9,10,11,12,13,14,15);
            accq[rt] = __builtin_amdgcn_wmma_f32_16x16x32_f16(
                false, a, false, bqf, (short)0, accq[rt], false, false);
            acck[rt] = __builtin_amdgcn_wmma_f32_16x16x32_f16(
                false, a, false, bkf, (short)0, acck[rt], false, false);
            accv[rt] = __builtin_amdgcn_wmma_f32_16x16x32_f16(
                false, a, false, bvf, (short)0, accv[rt], false, false);
        }
    }

    // ---- store results (C layout: VGPR r -> M = r + 8*(lane>=16)) ----
    const float biq = bQ[col], bik = bK[col], biv = bV[col];
    const int mbase = 8 * (lane >> 4);
    #pragma unroll
    for (int rt = 0; rt < 4; ++rt) {
        #pragma unroll
        for (int r = 0; r < 8; ++r) {
            size_t tok = (size_t)(rowbase + rt * 16 + mbase + r);
            size_t off = tok * EDIM + col;
            qws[off] = (_Float16)(accq[rt][r] + biq);
            kws[off] = (_Float16)(acck[rt][r] + bik);
            vws[off] = (_Float16)(accv[rt][r] + biv);
        }
    }
}

// ------------------------------------------------------------------
// Kernel 3: block-diagonal attention.  One wave per (8-token block, head).
// lane = (row n = lane>>2, 32-wide d-chunk dc = lane&3).
// ------------------------------------------------------------------
__global__ __launch_bounds__(256) void attn_kernel(
    const _Float16* __restrict__ qws, const _Float16* __restrict__ kws,
    const _Float16* __restrict__ vws, float* __restrict__ out)
{
    const int lane = threadIdx.x & 31;
    const int wid  = threadIdx.x >> 5;
    const int unit = blockIdx.x * 8 + wid;       // 16384 units
    const int head = unit & 1;
    const int blk  = unit >> 1;
    const size_t base = (size_t)blk * BSZ;
    const int n  = lane >> 2;
    const int dc = lane & 3;

    const _Float16* qp = qws + (base + n) * EDIM + head * HD + dc * 32;
    v8h q0 = *(const v8h*)(qp +  0);
    v8h q1 = *(const v8h*)(qp +  8);
    v8h q2 = *(const v8h*)(qp + 16);
    v8h q3 = *(const v8h*)(qp + 24);

    float s[8];
    #pragma unroll
    for (int m = 0; m < 8; ++m) {
        const _Float16* kp = kws + (base + m) * EDIM + head * HD + dc * 32;
        v8h k0 = *(const v8h*)(kp +  0);
        v8h k1 = *(const v8h*)(kp +  8);
        v8h k2 = *(const v8h*)(kp + 16);
        v8h k3 = *(const v8h*)(kp + 24);
        float acc = 0.f;
        #pragma unroll
        for (int j = 0; j < 8; ++j) {
            acc = fmaf((float)q0[j], (float)k0[j], acc);
            acc = fmaf((float)q1[j], (float)k1[j], acc);
            acc = fmaf((float)q2[j], (float)k2[j], acc);
            acc = fmaf((float)q3[j], (float)k3[j], acc);
        }
        s[m] = acc;
    }
    // reduce partial dots across the 4 lanes sharing row n (xor 1,2)
    #pragma unroll
    for (int m = 0; m < 8; ++m) {
        float t = s[m];
        t += __shfl_xor(t, 1, 32);
        t += __shfl_xor(t, 2, 32);
        s[m] = t * 0.08838834764831845f;   // 1/sqrt(128)
    }
    // softmax over the 8 in-block scores, fully in registers
    float mx = s[0];
    #pragma unroll
    for (int m = 1; m < 8; ++m) mx = fmaxf(mx, s[m]);
    float p[8], sum = 0.f;
    #pragma unroll
    for (int m = 0; m < 8; ++m) { p[m] = __expf(s[m] - mx); sum += p[m]; }
    const float inv = 1.f / sum;

    float o[32];
    #pragma unroll
    for (int i = 0; i < 32; ++i) o[i] = 0.f;
    #pragma unroll
    for (int m = 0; m < 8; ++m) {
        const _Float16* vp = vws + (base + m) * EDIM + head * HD + dc * 32;
        v8h v0 = *(const v8h*)(vp +  0);
        v8h v1 = *(const v8h*)(vp +  8);
        v8h v2 = *(const v8h*)(vp + 16);
        v8h v3 = *(const v8h*)(vp + 24);
        float pm = p[m];
        #pragma unroll
        for (int j = 0; j < 8; ++j) {
            o[j]      = fmaf(pm, (float)v0[j], o[j]);
            o[8 + j]  = fmaf(pm, (float)v1[j], o[8 + j]);
            o[16 + j] = fmaf(pm, (float)v2[j], o[16 + j]);
            o[24 + j] = fmaf(pm, (float)v3[j], o[24 + j]);
        }
    }
    float* op = out + (base + n) * EDIM + head * HD + dc * 32;
    #pragma unroll
    for (int i = 0; i < 8; ++i) {
        float4 f4 = make_float4(o[4*i] * inv, o[4*i+1] * inv,
                                o[4*i+2] * inv, o[4*i+3] * inv);
        ((float4*)op)[i] = f4;
    }
}

// ------------------------------------------------------------------
extern "C" void kernel_launch(void* const* d_in, const int* in_sizes, int n_in,
                              void* d_out, int out_size, void* d_ws, size_t ws_size,
                              hipStream_t stream) {
    const float* x  = (const float*)d_in[0];
    const float* Wq = (const float*)d_in[1];
    const float* bq = (const float*)d_in[2];
    const float* Wk = (const float*)d_in[3];
    const float* bk = (const float*)d_in[4];
    const float* Wv = (const float*)d_in[5];
    const float* bv = (const float*)d_in[6];
    const float* P  = (const float*)d_in[7];
    float* out = (float*)d_out;

    char* ws = (char*)d_ws;
    _Float16* WqT = (_Float16*)(ws + OFF_WQT);
    _Float16* WkT = (_Float16*)(ws + OFF_WKT);
    _Float16* WvT = (_Float16*)(ws + OFF_WVT);
    float*    bQ  = (float*)(ws + OFF_BQ);
    float*    bK  = (float*)(ws + OFF_BK);
    float*    bV  = (float*)(ws + OFF_BV);
    _Float16* qws = (_Float16*)(ws + OFF_Q);
    _Float16* kws = (_Float16*)(ws + OFF_K);
    _Float16* vws = (_Float16*)(ws + OFF_V);

    prep_kernel<<<772, 256, 0, stream>>>(Wq, bq, Wk, bk, Wv, bv, P,
                                         WqT, WkT, WvT, bQ, bK, bV);
    qkv_kernel<<<TOKENS / ROWS, 512, 0, stream>>>(x, WqT, WkT, WvT,
                                                  bQ, bK, bV, qws, kws, vws);
    attn_kernel<<<(TOKENS / BSZ) * 2 / 8, 256, 0, stream>>>(qws, kws, vws, out);
}
